// ClassifierModel_64931315581364
// MI455X (gfx1250) — matrix operand order repaired and checked by reading, not compiled, not run on hardware
//
#include <hip/hip_runtime.h>
#include <hip/hip_bf16.h>

// MI455X / gfx1250, wave32. Point MLPs + classifier via v_wmma_f32_16x16x32_f16,
// with TDM (tensor_load_to_lds) double-buffered input staging when available.

typedef __attribute__((ext_vector_type(16))) _Float16 v16h;
typedef __attribute__((ext_vector_type(8)))  _Float16 v8h;
typedef __attribute__((ext_vector_type(8)))  float    v8f;
typedef unsigned int u32x4 __attribute__((ext_vector_type(4)));
typedef int          i32x4 __attribute__((ext_vector_type(4)));
typedef int          i32x8 __attribute__((ext_vector_type(8)));

#define T_PTS          2097152
#define B_SEG          8192
#define TILES_PER_WAVE 8

#if defined(__has_builtin)
#if __has_builtin(__builtin_amdgcn_tensor_load_to_lds)
#define HAVE_TDM 1
#endif
#endif

__device__ __forceinline__ int lane_id() { return threadIdx.x & 31; }

#ifdef HAVE_TDM
// 1-row (1 x nElems) f32 tile load, global -> LDS, via the Tensor Data Mover.
// D# fields per CDNA5 ISA ch.8: group0 = {count/flags, lds_addr, global_addr, type},
// group1 = {data_size, tensor_dim0/1, tile_dim0/1/2, strides}. Groups 2/3 unused (2D).
__device__ __forceinline__ void tdm_load_f32_1d(unsigned lds_off,
                                                const float* gptr, int nElems) {
  const unsigned long long ga = (unsigned long long)(size_t)gptr;
  u32x4 g0;
  g0.x = 1u;                                     // count=1, user descriptor, load
  g0.y = lds_off;                                // lds_addr (bytes)
  g0.z = (unsigned)ga;                           // global_addr[31:0]
  g0.w = (unsigned)(ga >> 32) | (2u << 30);      // global_addr[56:32] | type=2
  i32x8 g1;
  g1[0] = 0x00020000;                            // data_size=2 (4B), no multicast
  g1[1] = (nElems & 0xffff) << 16;               // tensor_dim0[15:0]  (bits 63:48)
  g1[2] = ((nElems >> 16) & 0xffff) | (1 << 16); // tensor_dim0[31:16] | tensor_dim1=1
  g1[3] = (nElems & 0xffff) << 16;               // tile_dim0 = nElems (bits 127:112)
  g1[4] = 1;                                     // tile_dim1 = 1
  g1[5] = nElems;                                // tensor_dim0_stride[31:0]
  g1[6] = 0;
  g1[7] = 0;
  const i32x4 gz = {0, 0, 0, 0};
#if __clang_major__ >= 23
  const i32x8 gz8 = {0, 0, 0, 0, 0, 0, 0, 0};
  __builtin_amdgcn_tensor_load_to_lds(g0, g1, gz, gz, gz8, 0);
#else
  __builtin_amdgcn_tensor_load_to_lds(g0, g1, gz, gz, 0);
#endif
}
#endif

// B-matrix fragment for V_WMMA_F32_16X16X32_F16: B is 32x16 (KxN).
__device__ __forceinline__ v16h load_b_frag(const float* __restrict__ W, int ldn,
                                            int n, int Kvalid, int Nvalid, int k0) {
  const int lane = lane_id();
  const int kb = k0 + ((lane & 16) ? 16 : 0);
  v16h b;
#pragma unroll
  for (int h = 0; h < 16; ++h) {
    const int k = kb + h;
    float v = 0.0f;
    if (k < Kvalid && n < Nvalid) v = W[k * ldn + n];
    b[h] = (_Float16)v;
  }
  return b;
}

// A-matrix fragment (16x32 f16) from LDS row-major [16][ldk].
__device__ __forceinline__ v16h load_a_frag(const _Float16* L, int ldk, int k0) {
  const int lane = lane_id();
  const int m = lane & 15;
  const int kb = k0 + ((lane & 16) ? 16 : 0);
  const v8h lo = *(const v8h*)(L + m * ldk + kb);        // ds_load_b128
  const v8h hi = *(const v8h*)(L + m * ldk + kb + 8);    // ds_load_b128
  return __builtin_shufflevector(lo, hi, 0,1,2,3,4,5,6,7,8,9,10,11,12,13,14,15);
}

__device__ __forceinline__ v8f wmma_f16(v16h a, v16h b, v8f c) {
  return __builtin_amdgcn_wmma_f32_16x16x32_f16(false, a, false, b, (short)0, c,
                                                false, false);
}

// leaky_relu(x, 0.1) == max(x, 0.1*x)
__device__ __forceinline__ v8f lrelu8(v8f x) {
#pragma unroll
  for (int i = 0; i < 8; ++i) x[i] = fmaxf(x[i], 0.1f * x[i]);
  return x;
}

__device__ __forceinline__ v8f splat8(float v) {
  v8f c;
#pragma unroll
  for (int i = 0; i < 8; ++i) c[i] = v;
  return c;
}

__global__ void zero_kernel(float* __restrict__ p, int nfloats) {
  const int i = blockIdx.x * blockDim.x + threadIdx.x;
  if (i < nfloats) p[i] = 0.0f;
}

// Each wave loops over 8 tiles of 16 points; weight fragments loaded once.
// Inputs staged into LDS by double-buffered TDM DMA (overlaps the WMMA chain);
// segment accumulation is run-length reduced before global f32 atomics in L2.
__global__ __launch_bounds__(128)
void point_embed_kernel(const float* __restrict__ vel, const float* __restrict__ dist,
                        const int* __restrict__ seg,
                        const float* __restrict__ W1v, const float* __restrict__ b1v,
                        const float* __restrict__ W2v, const float* __restrict__ b2v,
                        const float* __restrict__ W1d, const float* __restrict__ b1d,
                        const float* __restrict__ W2d, const float* __restrict__ b2d,
                        float* __restrict__ sumV, float* __restrict__ sumD,
                        float* __restrict__ cnt) {
  __shared__ __align__(16) _Float16 lds[4][16 * 32];
#ifdef HAVE_TDM
  __shared__ __align__(16) float stage[4][2][80];   // [wave][buf][vel 32 | dist 48]
#endif
  const int wave = threadIdx.x >> 5;
  const int lane = lane_id();
  _Float16* L = lds[wave];
  const int waveBase = (blockIdx.x * 4 + wave) * (16 * TILES_PER_WAVE);

  // K=16..31 padding columns of the inter-layer LDS tile stay zero forever.
  for (int i = lane; i < 16 * 16; i += 32)
    L[(i >> 4) * 32 + 16 + (i & 15)] = (_Float16)0.0f;

  const int n = lane & 15;
  const int rowoff = (lane & 16) ? 8 : 0;

  // Weight fragments, hoisted out of the tile loop.
  const v16h Bv1  = load_b_frag(W1v, 16, n,       2, 16, 0);
  const v16h Bd1  = load_b_frag(W1d, 16, n,       3, 16, 0);
  const v16h Bv20 = load_b_frag(W2v, 32, n,      16, 32, 0);
  const v16h Bv21 = load_b_frag(W2v, 32, n + 16, 16, 32, 0);
  const v16h Bd20 = load_b_frag(W2d, 32, n,      16, 32, 0);
  const v16h Bd21 = load_b_frag(W2d, 32, n + 16, 16, 32, 0);
  const float bias1v = b1v[n], bias1d = b1d[n];
  const float bias2v0 = b2v[n], bias2v1 = b2v[n + 16];
  const float bias2d0 = b2d[n], bias2d1 = b2d[n + 16];

#ifdef HAVE_TDM
  const unsigned stOff[2] = { (unsigned)(size_t)&stage[wave][0][0],
                              (unsigned)(size_t)&stage[wave][1][0] };
  // Kick off tile 0's DMA (vel: 32 f32, dist: 48 f32).
  tdm_load_f32_1d(stOff[0],       vel  + (size_t)waveBase * 2, 32);
  tdm_load_f32_1d(stOff[0] + 128, dist + (size_t)waveBase * 3, 48);
#endif

  for (int t = 0; t < TILES_PER_WAVE; ++t) {
    const int base = waveBase + t * 16;
    const int buf  = t & 1;
    (void)buf;

#ifdef HAVE_TDM
    if (t + 1 < TILES_PER_WAVE) {                // prefetch next tile via TDM
      tdm_load_f32_1d(stOff[buf ^ 1],       vel  + (size_t)(base + 16) * 2, 32);
      tdm_load_f32_1d(stOff[buf ^ 1] + 128, dist + (size_t)(base + 16) * 3, 48);
      __builtin_amdgcn_s_wait_tensorcnt(2);      // tile t's 2 DMAs done (in-order)
    } else {
      __builtin_amdgcn_s_wait_tensorcnt(0);
    }
    const float* sv = &stage[wave][buf][0];
    const float* sd = &stage[wave][buf][32];
    asm volatile("" :: "r"(sv), "r"(sd) : "memory");  // TDM wrote LDS
#else
    if (t + 1 < TILES_PER_WAVE) {                // global_prefetch_b8
      __builtin_prefetch(vel  + (size_t)(base + 16 + lane) * 2, 0, 0);
      __builtin_prefetch(dist + (size_t)(base + 16 + lane) * 3, 0, 0);
    }
#endif

    // Segment ids for the 8 consecutive points (rows M = rowoff+i) this lane owns.
    int segs[8];
#pragma unroll
    for (int i = 0; i < 8; ++i) segs[i] = seg[base + rowoff + i];

    // ------------- velocity stream: 2 -> 16 -> 32 -------------
    v16h a;
#pragma unroll
    for (int i = 0; i < 16; ++i) a[i] = (_Float16)0.0f;
    if (lane < 16) {
#ifdef HAVE_TDM
      const float2 v = ((const float2*)sv)[lane];
#else
      const float2 v = *(const float2*)(vel + (size_t)(base + lane) * 2);
#endif
      a[0] = (_Float16)v.x;
      a[1] = (_Float16)v.y;
    }
    v8f h1 = lrelu8(wmma_f16(a, Bv1, splat8(bias1v)));
#pragma unroll
    for (int i = 0; i < 8; ++i)                  // D(16x16) -> LDS rows
      L[(rowoff + i) * 32 + n] = (_Float16)h1[i];
    const v16h a2 = load_a_frag(L, 32, 0);       // transpose via LDS
    const v8f v0 = lrelu8(wmma_f16(a2, Bv20, splat8(bias2v0)));
    const v8f v1 = lrelu8(wmma_f16(a2, Bv21, splat8(bias2v1)));

    // ------------- distance stream: 3 -> 16 -> 32 -------------
#pragma unroll
    for (int i = 0; i < 16; ++i) a[i] = (_Float16)0.0f;
    if (lane < 16) {
#ifdef HAVE_TDM
      const float* dp = sd + lane * 3;
#else
      const float* dp = dist + (size_t)(base + lane) * 3;
#endif
      a[0] = (_Float16)__expf(-dp[0] * 0.2f);
      a[1] = (_Float16)__expf(-dp[1] * 0.2f);
      a[2] = (_Float16)__expf(-dp[2] * 0.2f);
    }
    v8f h1d = lrelu8(wmma_f16(a, Bd1, splat8(bias1d)));
#pragma unroll
    for (int i = 0; i < 8; ++i)
      L[(rowoff + i) * 32 + n] = (_Float16)h1d[i];
    const v16h a2d = load_a_frag(L, 32, 0);
    const v8f d0 = lrelu8(wmma_f16(a2d, Bd20, splat8(bias2d0)));
    const v8f d1 = lrelu8(wmma_f16(a2d, Bd21, splat8(bias2d1)));

    // ------------- run-length reduced segment accumulation -------------
    int cur = segs[0];
    float aV0 = 0.f, aV1 = 0.f, aD0 = 0.f, aD1 = 0.f, ac = 0.f;
#pragma unroll
    for (int i = 0; i < 8; ++i) {
      if (segs[i] != cur) {
        const size_t o = (size_t)cur * 32 + n;
        atomicAdd(&sumV[o], aV0);      atomicAdd(&sumV[o + 16], aV1);
        atomicAdd(&sumD[o], aD0);      atomicAdd(&sumD[o + 16], aD1);
        if (n == 0) atomicAdd(&cnt[cur], ac);
        aV0 = aV1 = aD0 = aD1 = ac = 0.f;
        cur = segs[i];
      }
      aV0 += v0[i]; aV1 += v1[i]; aD0 += d0[i]; aD1 += d1[i]; ac += 1.0f;
    }
    const size_t o = (size_t)cur * 32 + n;
    atomicAdd(&sumV[o], aV0);      atomicAdd(&sumV[o + 16], aV1);
    atomicAdd(&sumD[o], aD0);      atomicAdd(&sumD[o + 16], aD1);
    if (n == 0) atomicAdd(&cnt[cur], ac);
  }
}

// One wave handles 16 segments through the 64->64->32->8 classifier + softmax.
__global__ __launch_bounds__(128)
void classifier_kernel(const float* __restrict__ sumV, const float* __restrict__ sumD,
                       const float* __restrict__ cnt,
                       const float* __restrict__ Wc1, const float* __restrict__ bc1,
                       const float* __restrict__ Wc2, const float* __restrict__ bc2,
                       const float* __restrict__ Wc3, const float* __restrict__ bc3,
                       float* __restrict__ out) {
  __shared__ __align__(16) _Float16 lds[4][16 * 64];
  const int wave = threadIdx.x >> 5;
  const int lane = lane_id();
  _Float16* L = lds[wave];
  const int segbase = (blockIdx.x * 4 + wave) * 16;
  const int n = lane & 15;
  const int rowoff = (lane & 16) ? 8 : 0;

  // Build h = [Vm | Dm] (mean-pooled) as f16 rows of 64 in LDS.
  for (int idx = lane; idx < 16 * 64; idx += 32) {
    const int r = idx >> 6, c = idx & 63;
    const int s = segbase + r;
    const float cn = fmaxf(cnt[s], 1.0f);
    const float v = (c < 32 ? sumV[(size_t)s * 32 + c]
                            : sumD[(size_t)s * 32 + (c - 32)]) / cn;
    L[idx] = (_Float16)v;
  }

  // Layer 1: 64 -> 64 (2 K-chunks x 4 N-tiles = 8 WMMA)
  v16h a0 = load_a_frag(L, 64, 0);
  v16h a1 = load_a_frag(L, 64, 32);
  v8f y[4];
#pragma unroll
  for (int nt = 0; nt < 4; ++nt) {
    const int col = n + 16 * nt;
    v8f acc = splat8(bc1[col]);
    acc = wmma_f16(a0, load_b_frag(Wc1, 64, col, 64, 64, 0),  acc);
    acc = wmma_f16(a1, load_b_frag(Wc1, 64, col, 64, 64, 32), acc);
    y[nt] = lrelu8(acc);
  }
#pragma unroll
  for (int nt = 0; nt < 4; ++nt)
#pragma unroll
    for (int i = 0; i < 8; ++i)
      L[(rowoff + i) * 64 + n + 16 * nt] = (_Float16)y[nt][i];

  // Layer 2: 64 -> 32 (2 K-chunks x 2 N-tiles = 4 WMMA)
  a0 = load_a_frag(L, 64, 0);
  a1 = load_a_frag(L, 64, 32);
  v8f z[2];
#pragma unroll
  for (int nt = 0; nt < 2; ++nt) {
    const int col = n + 16 * nt;
    v8f acc = splat8(bc2[col]);
    acc = wmma_f16(a0, load_b_frag(Wc2, 32, col, 64, 32, 0),  acc);
    acc = wmma_f16(a1, load_b_frag(Wc2, 32, col, 64, 32, 32), acc);
    z[nt] = lrelu8(acc);
  }
#pragma unroll
  for (int nt = 0; nt < 2; ++nt)
#pragma unroll
    for (int i = 0; i < 8; ++i)
      L[(rowoff + i) * 32 + n + 16 * nt] = (_Float16)z[nt][i];

  // Layer 3: 32 -> 8 (N padded to 16; 1 WMMA)
  const v16h a2 = load_a_frag(L, 32, 0);
  v8f s3 = splat8(n < 8 ? bc3[n] : 0.0f);
  s3 = wmma_f16(a2, load_b_frag(Wc3, 8, n, 32, 8, 0), s3);

  // Scores -> f32 LDS, then one lane per row does softmax + store.
  float* SF = (float*)L;
#pragma unroll
  for (int i = 0; i < 8; ++i) SF[(rowoff + i) * 16 + n] = s3[i];
  if (lane < 16) {
    float sc[8];
    float mx = -3.402823466e38f;
#pragma unroll
    for (int c = 0; c < 8; ++c) { sc[c] = SF[lane * 16 + c]; mx = fmaxf(mx, sc[c]); }
    float sum = 0.0f;
#pragma unroll
    for (int c = 0; c < 8; ++c) { sc[c] = __expf(sc[c] - mx); sum += sc[c]; }
    const float inv = 1.0f / sum;
    float* o = out + (size_t)(segbase + lane) * 8;
#pragma unroll
    for (int c = 0; c < 8; ++c) o[c] = sc[c] * inv;
  }
}

extern "C" void kernel_launch(void* const* d_in, const int* in_sizes, int n_in,
                              void* d_out, int out_size, void* d_ws, size_t ws_size,
                              hipStream_t stream) {
  const float* vel  = (const float*)d_in[0];
  const float* dist = (const float*)d_in[1];
  const int*   seg  = (const int*)d_in[2];
  // d_in[3] = num_segments scalar (hardcoded B_SEG = 8192)
  const float* W1v = (const float*)d_in[4];
  const float* b1v = (const float*)d_in[5];
  const float* W2v = (const float*)d_in[6];
  const float* b2v = (const float*)d_in[7];
  const float* W1d = (const float*)d_in[8];
  const float* b1d = (const float*)d_in[9];
  const float* W2d = (const float*)d_in[10];
  const float* b2d = (const float*)d_in[11];
  const float* Wc1 = (const float*)d_in[12];
  const float* bc1 = (const float*)d_in[13];
  const float* Wc2 = (const float*)d_in[14];
  const float* bc2 = (const float*)d_in[15];
  const float* Wc3 = (const float*)d_in[16];
  const float* bc3 = (const float*)d_in[17];

  float* sumV = (float*)d_ws;                    // [B_SEG][32]
  float* sumD = sumV + (size_t)B_SEG * 32;       // [B_SEG][32]
  float* cnt  = sumD + (size_t)B_SEG * 32;       // [B_SEG]

  const int nz = B_SEG * 65;                     // ~2.1 MB workspace, zero each call
  zero_kernel<<<(nz + 255) / 256, 256, 0, stream>>>((float*)d_ws, nz);

  // T / (16 pts * 4 waves * 8 tiles) = 4096 blocks -> exact cover.
  point_embed_kernel<<<T_PTS / (16 * 4 * TILES_PER_WAVE), 128, 0, stream>>>(
      vel, dist, seg, W1v, b1v, W2v, b2v, W1d, b1d, W2d, b2d, sumV, sumD, cnt);

  // B / 16 segment-tiles, 4 per block -> exact cover.
  classifier_kernel<<<B_SEG / (16 * 4), 128, 0, stream>>>(
      sumV, sumD, cnt, Wc1, bc1, Wc2, bc2, Wc3, bc3, (float*)d_out);
}
